// RegionProposalNetwork_56642028699790
// MI455X (gfx1250) — matrix operand order repaired and checked by reading, not compiled, not run on hardware
//
#include <hip/hip_runtime.h>
#include <hip/hip_bf16.h>
#include <math.h>

typedef __attribute__((ext_vector_type(16))) _Float16 v16h;
typedef __attribute__((ext_vector_type(8)))  _Float16 v8h;
typedef __attribute__((ext_vector_type(8)))  float    v8f;

#define NIMG 4
#define HH 64
#define WW 64
#define CIN 512
#define NANCH 36864           // 64*64*9
#define PRE_NMS 6000
#define POST_NMS 300
#define NMS_TH 0.7f
#define NEGV (-1.0e10f)

// ---- output element offsets (float32/int32 elements, concatenated flat) ----
#define OFF_LOCS    0                      // 4*36864*4 = 589824
#define OFF_SCORES  589824                 // 4*36864*2 = 294912
#define OFF_ROIS    884736                 // 1200*4    = 4800
#define OFF_ROIIDX  889536                 // 1200
#define OFF_ANCH    890736                 // 36864*4   = 147456

// ---- workspace byte offsets (all 256B aligned) ----
#define WS_FEAT16   ((size_t)0)                     // 4*66*66*512 f16 = 17,842,176 B
#define WS_W3       ((size_t)17842176)              // 9*512*512 f16   =  4,718,592 B
#define WS_WHEAD    ((size_t)22560768)              // 64*512 f16      =     65,536 B
#define WS_BHEAD    ((size_t)22626304)              // 64 f32
#define WS_MID16    ((size_t)22626560)              // 16384*512 f16   = 16,777,216 B
#define WS_FG       ((size_t)39403776)              // 4*36864 f32
#define WS_BOXES    ((size_t)39993600)              // 4*36864*4 f32
#define WS_SC       ((size_t)42352896)              // 4*36864 f32
#define WS_THR      ((size_t)42942720)              // 4 f32

// =====================================================================
// 1) base_feat (NCHW f32) -> padded NHWC f16 [4][66][66][512]
// =====================================================================
__global__ void k_prep_feat(const float* __restrict__ bf, _Float16* __restrict__ out) {
  int t = blockIdx.x * blockDim.x + threadIdx.x;
  int total = NIMG * 66 * 66 * CIN;
  if (t >= total) return;
  int ci = t & 511;
  int r  = t >> 9;
  int xx = r % 66; r /= 66;
  int yy = r % 66;
  int img = r / 66;
  float v = 0.0f;
  if (xx >= 1 && xx <= 64 && yy >= 1 && yy <= 64) {
    v = bf[(((size_t)img * CIN + ci) * HH + (yy - 1)) * WW + (xx - 1)];
  }
  out[t] = (_Float16)v;
}

// =====================================================================
// 2) W_conv1 (co,ci,3,3) f32 -> [kk][co][ci] f16
// =====================================================================
__global__ void k_prep_w3(const float* __restrict__ w, _Float16* __restrict__ out) {
  int t = blockIdx.x * blockDim.x + threadIdx.x;
  int total = 9 * CIN * CIN;
  if (t >= total) return;
  int ci = t & 511;
  int r  = t >> 9;
  int co = r % CIN;
  int kk = r / CIN;
  out[t] = (_Float16)w[((size_t)co * CIN + ci) * 9 + kk];
}

// =====================================================================
// 3) head weights: rows 0..35 = W_loc, 36..53 = W_score, 54..63 = 0
// =====================================================================
__global__ void k_prep_whead(const float* __restrict__ wloc, const float* __restrict__ wsc,
                             const float* __restrict__ bloc, const float* __restrict__ bsc,
                             _Float16* __restrict__ wh, float* __restrict__ bh) {
  int t = blockIdx.x * blockDim.x + threadIdx.x;
  if (t >= 64 * CIN) return;
  int ci = t & 511;
  int c  = t >> 9;
  float v = 0.0f;
  if (c < 36)       v = wloc[(size_t)c * CIN + ci];
  else if (c < 54)  v = wsc[(size_t)(c - 36) * CIN + ci];
  wh[t] = (_Float16)v;
  if (t < 64) {
    float b = 0.0f;
    if (t < 36)      b = bloc[t];
    else if (t < 54) b = bsc[t - 36];
    bh[t] = b;
  }
}

// =====================================================================
// 4) 3x3 conv 512->512 + bias + ReLU via WMMA implicit GEMM.
//    Wave tile: 32(m) x 64(n), 8 accumulators, K = 9*512 in steps of 32.
//    12 b128 loads per 8 WMMAs (1.5 loads/WMMA).
//    A frag (16x32 f16): lane l holds M=l%16, K runs [ka0,ka0+8) & [ka0+16,ka0+24), ka0=(l>=16)*8
//    B frag (32x16 f16): lane l holds N=l%16, K runs [kb0,kb0+8) & [kb0+16,kb0+24), kb0=(l>=16)*8
//    C/D (16x16 f32):    lane l holds N=l%16, M=v+(l>=16)*8 for VGPR v
// =====================================================================
__global__ void __launch_bounds__(256) k_conv3x3(const _Float16* __restrict__ feat,
                                                 const _Float16* __restrict__ w3,
                                                 const float* __restrict__ bconv,
                                                 _Float16* __restrict__ mid) {
  const int wave  = blockIdx.x * 8 + (threadIdx.x >> 5);
  const int lane  = threadIdx.x & 31;
  const int mtile = wave >> 3;          // 0..511
  const int ntile = wave & 7;           // 0..7
  const int nbase = ntile * 64;
  const int mbase = mtile * 32;
  const int img   = mbase >> 12;
  const int rem   = mbase & 4095;
  const int y     = rem >> 6;
  const int x0    = rem & 63;           // 0 or 32
  const int mm    = lane & 15;
  const int hi    = lane >> 4;

  v8f acc[8];
#pragma unroll
  for (int j = 0; j < 8; ++j) acc[j] = (v8f)0.0f;

  for (int kk = 0; kk < 9; ++kk) {
    const int ky = kk / 3, kx = kk % 3;
    const _Float16* arow0 = feat + (((size_t)img * 66 + (y + ky)) * 66 + (x0 + mm + kx)) * CIN;
    const _Float16* arow1 = arow0 + (size_t)16 * CIN;
    const _Float16* wkk   = w3 + (size_t)kk * CIN * CIN;
#pragma unroll 2
    for (int ci0 = 0; ci0 < CIN; ci0 += 32) {
      v8h a00 = *(const v8h*)(arow0 + ci0 + hi * 8);
      v8h a01 = *(const v8h*)(arow0 + ci0 + hi * 8 + 16);
      v16h A0 = __builtin_shufflevector(a00, a01, 0,1,2,3,4,5,6,7,8,9,10,11,12,13,14,15);
      v8h a10 = *(const v8h*)(arow1 + ci0 + hi * 8);
      v8h a11 = *(const v8h*)(arow1 + ci0 + hi * 8 + 16);
      v16h A1 = __builtin_shufflevector(a10, a11, 0,1,2,3,4,5,6,7,8,9,10,11,12,13,14,15);
#pragma unroll
      for (int j = 0; j < 4; ++j) {
        const _Float16* bp = wkk + (size_t)(nbase + 16 * j + mm) * CIN + ci0 + hi * 8;
        v8h b0 = *(const v8h*)(bp);
        v8h b1 = *(const v8h*)(bp + 16);
        v16h B = __builtin_shufflevector(b0, b1, 0,1,2,3,4,5,6,7,8,9,10,11,12,13,14,15);
        acc[2 * j]     = __builtin_amdgcn_wmma_f32_16x16x32_f16(false, A0, false, B,
                                                                (short)0, acc[2 * j], false, false);
        acc[2 * j + 1] = __builtin_amdgcn_wmma_f32_16x16x32_f16(false, A1, false, B,
                                                                (short)0, acc[2 * j + 1], false, false);
      }
    }
  }

#pragma unroll
  for (int j = 0; j < 4; ++j) {
    const int n = nbase + 16 * j + mm;
    const float b = bconv[n];
#pragma unroll
    for (int h = 0; h < 2; ++h) {
#pragma unroll
      for (int v = 0; v < 8; ++v) {
        const int m = mbase + h * 16 + v + hi * 8;
        float val = acc[2 * j + h][v] + b;
        mid[(size_t)m * CIN + n] = (_Float16)(val > 0.0f ? val : 0.0f);
      }
    }
  }
}

// =====================================================================
// 5) fused 1x1 heads: GEMM [16384 x 64] = mid16[16384 x 512] * Whead^T
//    epilogue writes rpn_locs (c<36) and rpn_scores (36<=c<54) directly.
// =====================================================================
__global__ void __launch_bounds__(256) k_head(const _Float16* __restrict__ mid,
                                              const _Float16* __restrict__ wh,
                                              const float* __restrict__ bh,
                                              float* __restrict__ out_locs,
                                              float* __restrict__ out_scores) {
  const int wave  = blockIdx.x * 8 + (threadIdx.x >> 5);   // 0..1023 == mtile
  const int lane  = threadIdx.x & 31;
  const int mbase = wave * 16;
  const int mm    = lane & 15;
  const int hi    = lane >> 4;

  v8f acc[4];
#pragma unroll
  for (int j = 0; j < 4; ++j) acc[j] = (v8f)0.0f;

  const _Float16* arow = mid + (size_t)(mbase + mm) * CIN;
#pragma unroll 4
  for (int ci0 = 0; ci0 < CIN; ci0 += 32) {
    v8h a0 = *(const v8h*)(arow + ci0 + hi * 8);
    v8h a1 = *(const v8h*)(arow + ci0 + hi * 8 + 16);
    v16h A = __builtin_shufflevector(a0, a1, 0,1,2,3,4,5,6,7,8,9,10,11,12,13,14,15);
#pragma unroll
    for (int j = 0; j < 4; ++j) {
      const _Float16* bp = wh + (size_t)(16 * j + mm) * CIN + ci0 + hi * 8;
      v8h b0 = *(const v8h*)(bp);
      v8h b1 = *(const v8h*)(bp + 16);
      v16h B = __builtin_shufflevector(b0, b1, 0,1,2,3,4,5,6,7,8,9,10,11,12,13,14,15);
      acc[j] = __builtin_amdgcn_wmma_f32_16x16x32_f16(false, A, false, B,
                                                      (short)0, acc[j], false, false);
    }
  }

#pragma unroll
  for (int j = 0; j < 4; ++j) {
    const int c = 16 * j + mm;
    const float b = bh[c];
#pragma unroll
    for (int v = 0; v < 8; ++v) {
      const int m = mbase + v + hi * 8;
      float val = acc[j][v] + b;
      if (c < 36)       out_locs[(size_t)m * 36 + c] = val;
      else if (c < 54)  out_scores[(size_t)m * 18 + (c - 36)] = val;
    }
  }
}

// =====================================================================
// 6) anchors [36864][4]
// =====================================================================
__global__ void k_anchors(float* __restrict__ out) {
  int i = blockIdx.x * blockDim.x + threadIdx.x;
  if (i >= NANCH) return;
  int pix = i / 9, a = i % 9;
  const float ratios[3] = {0.5f, 1.0f, 2.0f};
  const float scales[3] = {8.0f, 16.0f, 32.0f};
  float r = ratios[a / 3], s = scales[a % 3];
  float h = 16.0f * s * sqrtf(r);
  float w = 16.0f * s * sqrtf(1.0f / r);
  float cy = (float)(pix / 64) * 16.0f + 8.0f;
  float cx = (float)(pix % 64) * 16.0f + 8.0f;
  out[i * 4 + 0] = cy - 0.5f * h;
  out[i * 4 + 1] = cx - 0.5f * w;
  out[i * 4 + 2] = cy + 0.5f * h;
  out[i * 4 + 3] = cx + 0.5f * w;
}

// =====================================================================
// 7) fg = softmax(scores)[...,1]
// =====================================================================
__global__ void k_fg(const float* __restrict__ scores, float* __restrict__ fg) {
  int t = blockIdx.x * blockDim.x + threadIdx.x;
  if (t >= NIMG * NANCH) return;
  int img = t / NANCH;
  int rest = t % NANCH;
  int pix = rest / 9, a = rest % 9;
  size_t m = (size_t)img * 4096 + pix;
  float s0 = scores[m * 18 + 2 * a];
  float s1 = scores[m * 18 + 2 * a + 1];
  fg[t] = 1.0f / (1.0f + __expf(s0 - s1));
}

// =====================================================================
// 8) decode boxes, clip, min-size filter -> boxes + masked scores
// =====================================================================
__global__ void k_boxes(const float* __restrict__ anch, const float* __restrict__ locs,
                        const float* __restrict__ fg, float* __restrict__ boxes,
                        float* __restrict__ sc, const int* __restrict__ ph,
                        const int* __restrict__ pw) {
  int t = blockIdx.x * blockDim.x + threadIdx.x;
  if (t >= NIMG * NANCH) return;
  int j = t % NANCH;
  float ih = (float)ph[0], iw = (float)pw[0];
  float a0 = anch[j * 4 + 0], a1 = anch[j * 4 + 1], a2 = anch[j * 4 + 2], a3 = anch[j * 4 + 3];
  float hh = a2 - a0, ww = a3 - a1;
  float cy = a0 + 0.5f * hh, cx = a1 + 0.5f * ww;
  float dy = locs[(size_t)t * 4 + 0], dx = locs[(size_t)t * 4 + 1];
  float dh = locs[(size_t)t * 4 + 2], dw = locs[(size_t)t * 4 + 3];
  float ncy = dy * hh + cy, ncx = dx * ww + cx;
  float nh = __expf(dh) * hh, nw = __expf(dw) * ww;
  float b0 = fminf(fmaxf(ncy - 0.5f * nh, 0.0f), ih);
  float b1 = fminf(fmaxf(ncx - 0.5f * nw, 0.0f), iw);
  float b2 = fminf(fmaxf(ncy + 0.5f * nh, 0.0f), ih);
  float b3 = fminf(fmaxf(ncx + 0.5f * nw, 0.0f), iw);
  boxes[(size_t)t * 4 + 0] = b0;
  boxes[(size_t)t * 4 + 1] = b1;
  boxes[(size_t)t * 4 + 2] = b2;
  boxes[(size_t)t * 4 + 3] = b3;
  bool ok = ((b2 - b0) >= 16.0f) && ((b3 - b1) >= 16.0f);
  sc[t] = ok ? fg[t] : NEGV;
}

// =====================================================================
// 9) per-image histogram threshold approximating top-PRE_NMS
// =====================================================================
__global__ void __launch_bounds__(1024) k_hist(const float* __restrict__ sc,
                                               float* __restrict__ thr) {
  __shared__ unsigned int hist[1024];
  int img = blockIdx.x, tid = threadIdx.x;
  hist[tid] = 0u;
  __syncthreads();
  const float* s = sc + (size_t)img * NANCH;
  for (int j = tid; j < NANCH; j += 1024) {
    float v = s[j];
    if (v > NEGV * 0.5f) {
      int b = (int)(v * 1024.0f);
      b = b < 0 ? 0 : (b > 1023 ? 1023 : b);
      atomicAdd(&hist[b], 1u);
    }
  }
  __syncthreads();
  if (tid == 0) {
    unsigned int cum = 0;
    float t = -3.0e38f;
    for (int b = 1023; b >= 0; --b) {
      cum += hist[b];
      if (cum >= PRE_NMS) { t = (float)b / 1024.0f; break; }
    }
    thr[img] = t;
  }
}

// =====================================================================
// 10) NMS: one 1024-thread workgroup per image; all 36864 scores live in
//     LDS (147KB — only possible with CDNA5's 320KB/WGP LDS). The initial
//     global->LDS staging uses GLOBAL_LOAD_ASYNC_TO_LDS_B128 (ASYNCcnt
//     path): kernel has no static LDS, so dynamic-shared offset == 0 and
//     the LDS destination address for s[j] is exactly j*4 bytes.
// =====================================================================
__global__ void __launch_bounds__(1024) k_nms(const float* __restrict__ boxes,
                                              const float* __restrict__ sc,
                                              const float* __restrict__ thr,
                                              float* __restrict__ rois,
                                              int* __restrict__ roi_idx) {
  extern __shared__ char smem[];
  float* s   = (float*)smem;                       // 36864 floats (offset 0)
  float* rb  = (float*)(smem + NANCH * 4);         // 1024
  int*   ri  = (int*)(smem + NANCH * 4 + 4096);    // 1024
  float* bb  = (float*)(smem + NANCH * 4 + 8192);  // 4

  const int img = blockIdx.x, tid = threadIdx.x;
  const float* B = boxes + (size_t)img * NANCH * 4;
  const float t0 = thr[img];
  const float* src = sc + (size_t)img * NANCH;

  // ---- async copy scores -> LDS (9 x b128 per thread) ----
  for (int j = tid * 4; j < NANCH; j += 4096) {
    unsigned int ldsoff = (unsigned int)(j * 4);   // byte offset in LDS
    unsigned long long gaddr = (unsigned long long)(const void*)(src + j);
    asm volatile("global_load_async_to_lds_b128 %0, %1, off"
                 :: "v"(ldsoff), "v"(gaddr) : "memory");
  }
  asm volatile("s_wait_asynccnt 0x0" ::: "memory");
  __syncthreads();
  // ---- apply PRE_NMS threshold in place ----
  for (int j = tid; j < NANCH; j += 1024) {
    float v = s[j];
    s[j] = (v >= t0) ? v : NEGV;
  }
  __syncthreads();

  for (int step = 0; step < POST_NMS; ++step) {
    // ---- argmax (first-occurrence tie-break) ----
    float bestv = -3.4e38f;
    int besti = 0;
    for (int j = tid; j < NANCH; j += 1024) {
      float v = s[j];
      if (v > bestv || (v == bestv && j < besti)) { bestv = v; besti = j; }
    }
    rb[tid] = bestv; ri[tid] = besti;
    __syncthreads();
    for (int off = 512; off > 0; off >>= 1) {
      if (tid < off) {
        float ov = rb[tid + off]; int oi = ri[tid + off];
        if (ov > rb[tid] || (ov == rb[tid] && oi < ri[tid])) { rb[tid] = ov; ri[tid] = oi; }
      }
      __syncthreads();
    }
    if (tid == 0) {
      int i = ri[0];
      float v = rb[0];
      float x0 = B[(size_t)i * 4 + 0], x1 = B[(size_t)i * 4 + 1];
      float x2 = B[(size_t)i * 4 + 2], x3 = B[(size_t)i * 4 + 3];
      bb[0] = x0; bb[1] = x1; bb[2] = x2; bb[3] = x3;
      float valid = (v > NEGV * 0.5f) ? 1.0f : 0.0f;
      size_t o = ((size_t)img * POST_NMS + step) * 4;
      rois[o + 0] = x0 * valid; rois[o + 1] = x1 * valid;
      rois[o + 2] = x2 * valid; rois[o + 3] = x3 * valid;
      roi_idx[img * POST_NMS + step] = img;
    }
    __syncthreads();
    // ---- suppress ----
    float y0 = bb[0], y1 = bb[1], y2 = bb[2], y3 = bb[3];
    float a1 = (y2 - y0) * (y3 - y1);
    for (int j = tid; j < NANCH; j += 1024) {
      if (s[j] <= NEGV) continue;
      float z0 = B[(size_t)j * 4 + 0], z1 = B[(size_t)j * 4 + 1];
      float z2 = B[(size_t)j * 4 + 2], z3 = B[(size_t)j * 4 + 3];
      float ty = fmaxf(y0, z0), tx = fmaxf(y1, z1);
      float by = fminf(y2, z2), bx = fminf(y3, z3);
      float inter = fmaxf(by - ty, 0.0f) * fmaxf(bx - tx, 0.0f);
      float a2 = (z2 - z0) * (z3 - z1);
      float iou = inter / (a1 + a2 - inter + 1e-9f);
      if (iou > NMS_TH) s[j] = NEGV;
    }
    __syncthreads();
  }
}

// =====================================================================
extern "C" void kernel_launch(void* const* d_in, const int* in_sizes, int n_in,
                              void* d_out, int out_size, void* d_ws, size_t ws_size,
                              hipStream_t stream) {
  const float* base_feat = (const float*)d_in[0];
  const float* W_conv1   = (const float*)d_in[1];
  const float* b_conv1   = (const float*)d_in[2];
  const float* W_score   = (const float*)d_in[3];
  const float* b_score   = (const float*)d_in[4];
  const float* W_loc     = (const float*)d_in[5];
  const float* b_loc     = (const float*)d_in[6];
  const int*   img_h     = (const int*)d_in[7];
  const int*   img_w     = (const int*)d_in[8];

  float* out        = (float*)d_out;
  float* out_locs   = out + OFF_LOCS;
  float* out_scores = out + OFF_SCORES;
  float* out_rois   = out + OFF_ROIS;
  int*   out_ridx   = (int*)(out + OFF_ROIIDX);
  float* out_anch   = out + OFF_ANCH;

  char* ws = (char*)d_ws;
  _Float16* feat16 = (_Float16*)(ws + WS_FEAT16);
  _Float16* w3     = (_Float16*)(ws + WS_W3);
  _Float16* whead  = (_Float16*)(ws + WS_WHEAD);
  float*    bhead  = (float*)(ws + WS_BHEAD);
  _Float16* mid16  = (_Float16*)(ws + WS_MID16);
  float*    fgbuf  = (float*)(ws + WS_FG);
  float*    boxes  = (float*)(ws + WS_BOXES);
  float*    scbuf  = (float*)(ws + WS_SC);
  float*    thrbuf = (float*)(ws + WS_THR);

  k_prep_feat<<<(NIMG * 66 * 66 * CIN) / 256, 256, 0, stream>>>(base_feat, feat16);
  k_prep_w3<<<(9 * CIN * CIN) / 256, 256, 0, stream>>>(W_conv1, w3);
  k_prep_whead<<<(64 * CIN) / 256, 256, 0, stream>>>(W_loc, W_score, b_loc, b_score, whead, bhead);

  k_conv3x3<<<512, 256, 0, stream>>>(feat16, w3, b_conv1, mid16);
  k_head<<<128, 256, 0, stream>>>(mid16, whead, bhead, out_locs, out_scores);

  k_anchors<<<NANCH / 256, 256, 0, stream>>>(out_anch);
  k_fg<<<(NIMG * NANCH) / 256, 256, 0, stream>>>(out_scores, fgbuf);
  k_boxes<<<(NIMG * NANCH) / 256, 256, 0, stream>>>(out_anch, out_locs, fgbuf,
                                                    boxes, scbuf, img_h, img_w);
  k_hist<<<NIMG, 1024, 0, stream>>>(scbuf, thrbuf);

  size_t nms_smem = (size_t)NANCH * 4 + 4096 + 4096 + 64;
  k_nms<<<NIMG, 1024, nms_smem, stream>>>(boxes, scbuf, thrbuf, out_rois, out_ridx);

  (void)in_sizes; (void)n_in; (void)out_size; (void)ws_size;
}
